// LSTMLanguageModel_55190329753765
// MI455X (gfx1250) — matrix operand order repaired and checked by reading, not compiled, not run on hardware
//
#include <hip/hip_runtime.h>
#include <hip/hip_bf16.h>

#define VOCAB 32000
#define HID   1024
#define BATCH 8
#define SEQ   512
#define G4H   (4 * HID)
#define BS    (BATCH * SEQ)

typedef __attribute__((ext_vector_type(16))) __bf16 v16bf;
typedef __attribute__((ext_vector_type(8)))  __bf16 v8bf;
typedef __attribute__((ext_vector_type(4)))  __bf16 v4bf;
typedef __attribute__((ext_vector_type(8)))  float  v8f;

// ---------------------------------------------------------------------------
// bf16 GEMM with WMMA. Requirements of this specialized kernel:
//   * lda == K  (A tile rows contiguous -> 16xK tile is one flat 32KB block)
//   * K <= 1024, K % 32 == 0, N % 256 == 0, M % 16 == 0 (M=16 pad for scan)
// Block = 128 threads = 4 wave32. Each wave computes a 16x64 strip.
// A tile is staged once per workgroup into LDS via async global->LDS DMA
// (ASYNCcnt), then fragments are read with ds_load_b128. B streams from
// global (L2/HBM) with pointer-bumped per-element 16-bit loads + prefetch.
//
// Fragment layouts per CDNA5 ISA 7.12.2 (wave32):
//   A (16x32): lane<16 row=lane, K={k0..k0+7, k0+16..k0+23}; lane>=16 +8.
//   B (32x16): lane holds col N=lane&15, K={k0..k0+7, k0+16..k0+23}; lane>=16 +8 (same as A).
//   C/D 16x16: VGPR r, lane L -> M = r + 8*(L>=16), N = L&15.
// ---------------------------------------------------------------------------
__global__ __launch_bounds__(128) void gemm_bf16_wmma(
    const __bf16* __restrict__ A, const __bf16* __restrict__ B,
    const float* __restrict__ bias, float* __restrict__ C,
    int M, int N, int K, int ldb, int ldc)
{
  __shared__ __bf16 Ash[16 * 1024];

  unsigned tid  = threadIdx.x;
  unsigned lane = tid & 31u;
  unsigned wave = tid >> 5;
  int m0 = blockIdx.y * 16;
  int n0 = (blockIdx.x * 4 + (int)wave) * 64;
  (void)M;

  // ---- Stage the contiguous 16xK A tile into LDS with async DMA ----------
  {
    const __bf16* Atile = A + (size_t)m0 * (size_t)K;
    int chunks = (16 * K) / 8;  // 16-byte chunks
    for (int i = (int)tid; i < chunks; i += 128) {
      unsigned lds_off = (unsigned)(size_t)(&Ash[(size_t)i * 8]);
      unsigned long long ga = (unsigned long long)(size_t)(Atile + (size_t)i * 8);
      asm volatile("global_load_async_to_lds_b128 %0, %1, off"
                   :: "v"(lds_off), "v"(ga) : "memory");
    }
    asm volatile("s_wait_asynccnt 0x0" ::: "memory");
    __syncthreads();
  }

  // ---- Loop-invariant per-lane addressing (no per-iteration multiplies) --
  unsigned row  = lane & 15u;
  unsigned kb   = (lane >> 4) * 8u;
  const __bf16* arow = &Ash[(size_t)row * (size_t)K + kb];

  unsigned krow = (lane >> 4) * 8u;
  const __bf16* bp = B + (size_t)krow * (size_t)ldb + (size_t)n0 + (size_t)(lane & 15u);
  const size_t bstep = (size_t)32 * (size_t)ldb;

  v8f acc0 = {}, acc1 = {}, acc2 = {}, acc3 = {};

  for (int k0 = 0; k0 < K; k0 += 32) {
    v8bf lo = *(const v8bf*)(arow);        // LDS: K = k0+kb .. +7
    v8bf hi = *(const v8bf*)(arow + 16);   // LDS: K = k0+kb+16 .. +23
    v16bf a = __builtin_shufflevector(lo, hi,
              0,1,2,3,4,5,6,7,8,9,10,11,12,13,14,15);

    v16bf b0, b1, b2, b3;
#pragma unroll
    for (int s = 0; s < 16; ++s) {
      size_t ko = (size_t)(s < 8 ? s : s + 8) * (size_t)ldb;
      b0[s] = bp[ko];
      b1[s] = bp[ko + 16];
      b2[s] = bp[ko + 32];
      b3[s] = bp[ko + 48];
    }

    if (k0 + 32 < K) __builtin_prefetch(bp + bstep, 0, 1);

    acc0 = __builtin_amdgcn_wmma_f32_16x16x32_bf16(false, a, false, b0, (short)0, acc0, false, false);
    acc1 = __builtin_amdgcn_wmma_f32_16x16x32_bf16(false, a, false, b1, (short)0, acc1, false, false);
    acc2 = __builtin_amdgcn_wmma_f32_16x16x32_bf16(false, a, false, b2, (short)0, acc2, false, false);
    acc3 = __builtin_amdgcn_wmma_f32_16x16x32_bf16(false, a, false, b3, (short)0, acc3, false, false);

    arow += 32;
    bp   += bstep;
  }

  // ---- Epilogue: bias + store ------------------------------------------
  unsigned rsel = (lane >> 4) * 8u;
  unsigned col  = lane & 15u;
  float bv0 = 0.f, bv1 = 0.f, bv2 = 0.f, bv3 = 0.f;
  if (bias) {
    bv0 = bias[n0 +  0 + col];
    bv1 = bias[n0 + 16 + col];
    bv2 = bias[n0 + 32 + col];
    bv3 = bias[n0 + 48 + col];
  }
#pragma unroll
  for (int r = 0; r < 8; ++r) {
    size_t m = (size_t)(m0 + r + (int)rsel);
    float* crow = C + m * (size_t)ldc + n0;
    crow[ 0 + col] = acc0[r] + bv0;
    crow[16 + col] = acc1[r] + bv1;
    crow[32 + col] = acc2[r] + bv2;
    crow[48 + col] = acc3[r] + bv3;
  }
}

// ---------------------------------------------------------------------------
__global__ void f32_to_bf16_kernel(const float* __restrict__ in,
                                   __bf16* __restrict__ out, int n) {
  int i = (blockIdx.x * blockDim.x + threadIdx.x) * 4;  // n % 4 == 0 here
  if (i < n) {
    float4 f = *(const float4*)(in + i);
    v4bf o;
    o[0] = (__bf16)f.x; o[1] = (__bf16)f.y;
    o[2] = (__bf16)f.z; o[3] = (__bf16)f.w;
    *(v4bf*)(out + i) = o;
  }
}

// emb[r, h] = w_out_bf16[h, ids[r]]  (weight-tied embedding gather)
__global__ void embed_gather_kernel(const __bf16* __restrict__ w_out_bf,
                                    const int* __restrict__ ids,
                                    __bf16* __restrict__ emb) {
  int h = blockIdx.x * blockDim.x + threadIdx.x;  // 0..HID-1
  int r = blockIdx.y;                             // 0..BS-1
  emb[(size_t)r * HID + h] = w_out_bf[(size_t)h * VOCAB + ids[r]];
}

// zero padded-h (16 x HID bf16) and cell state (8 x HID f32)
__global__ void init_state_kernel(__bf16* __restrict__ hbf, float* __restrict__ c) {
  int i = blockIdx.x * blockDim.x + threadIdx.x;  // 0..16*HID-1
  if (i < 16 * HID) hbf[i] = (__bf16)0.0f;
  if (i < BATCH * HID) c[i] = 0.0f;
}

__device__ __forceinline__ float sigmoidf_(float x) {
  return 1.0f / (1.0f + __expf(-x));
}

// pointwise LSTM cell for timestep t
__global__ void lstm_cell_kernel(const float* __restrict__ xg,     // [BS, 4H]
                                 const float* __restrict__ gates,  // [16, 4H], rows 0..7 valid
                                 float* __restrict__ c,            // [8, H]
                                 __bf16* __restrict__ hbf,         // [16, H]
                                 __bf16* __restrict__ hs,          // [BS, H]
                                 int t) {
  int idx = blockIdx.x * blockDim.x + threadIdx.x;  // 0..BATCH*HID-1
  int b = idx >> 10;
  int j = idx & (HID - 1);
  size_t xr = ((size_t)b * SEQ + (size_t)t) * G4H;
  size_t gr = (size_t)b * G4H;
  float gi = xg[xr +          j] + gates[gr +          j];
  float gf = xg[xr +   HID +  j] + gates[gr +   HID +  j];
  float gg = xg[xr + 2*HID +  j] + gates[gr + 2*HID +  j];
  float go = xg[xr + 3*HID +  j] + gates[gr + 3*HID +  j];
  float i_ = sigmoidf_(gi);
  float f_ = sigmoidf_(gf);
  float g_ = tanhf(gg);
  float o_ = sigmoidf_(go);
  float cn = f_ * c[idx] + i_ * g_;
  float hn = o_ * tanhf(cn);
  c[idx] = cn;
  __bf16 hb = (__bf16)hn;
  hbf[(size_t)b * HID + j] = hb;
  hs[((size_t)b * SEQ + (size_t)t) * HID + j] = hb;
}

// ---------------------------------------------------------------------------
extern "C" void kernel_launch(void* const* d_in, const int* in_sizes, int n_in,
                              void* d_out, int out_size, void* d_ws, size_t ws_size,
                              hipStream_t stream) {
  const int*   ids    = (const int*)  d_in[0];  // [8,512]
  const float* w_out  = (const float*)d_in[1];  // [1024,32000]
  const float* b_out  = (const float*)d_in[2];  // [32000]
  const float* W_ih   = (const float*)d_in[3];  // [1024,4096]
  const float* W_hh   = (const float*)d_in[4];  // [1024,4096]
  const float* b_lstm = (const float*)d_in[5];  // [4096]
  float* logits = (float*)d_out;                // [8,512,32000]

  // Workspace carve-out (256B aligned), ~167 MB total
  char* w = (char*)d_ws;
  size_t off = 0;
  auto carve = [&](size_t bytes) -> char* {
    char* p = w + off;
    off = (off + bytes + 255) & ~(size_t)255;
    return p;
  };
  __bf16* wih_bf  = (__bf16*)carve((size_t)HID * G4H * 2);       //  8 MiB
  __bf16* whh_bf  = (__bf16*)carve((size_t)HID * G4H * 2);       //  8 MiB
  __bf16* wout_bf = (__bf16*)carve((size_t)HID * VOCAB * 2);     // 62.5 MiB
  __bf16* emb_bf  = (__bf16*)carve((size_t)BS * HID * 2);        //  8 MiB
  float*  xgates  = (float*) carve((size_t)BS * G4H * 4);        // 64 MiB
  __bf16* h_bf    = (__bf16*)carve((size_t)16 * HID * 2);        // 32 KiB (padded 8->16)
  float*  c_st    = (float*) carve((size_t)BATCH * HID * 4);     // 32 KiB
  float*  gates   = (float*) carve((size_t)16 * G4H * 4);        // 256 KiB
  __bf16* hs_bf   = (__bf16*)carve((size_t)BS * HID * 2);        //  8 MiB
  (void)ws_size; (void)n_in; (void)in_sizes; (void)out_size;

  // 1) Weight conversion f32 -> bf16 (one pass; amortized over whole model)
  {
    int n1 = HID * G4H;
    f32_to_bf16_kernel<<<(n1 / 4 + 255) / 256, 256, 0, stream>>>(W_ih, wih_bf, n1);
    f32_to_bf16_kernel<<<(n1 / 4 + 255) / 256, 256, 0, stream>>>(W_hh, whh_bf, n1);
    int n2 = HID * VOCAB;
    f32_to_bf16_kernel<<<(n2 / 4 + 255) / 256, 256, 0, stream>>>(w_out, wout_bf, n2);
  }

  // 2) Weight-tied embedding gather (strided column reads served by L2)
  embed_gather_kernel<<<dim3(HID / 256, BS), 256, 0, stream>>>(wout_bf, ids, emb_bf);

  // 3) x_gates = emb @ W_ih + b_lstm   [4096,1024]@[1024,4096]
  gemm_bf16_wmma<<<dim3(G4H / 256, BS / 16), 128, 0, stream>>>(
      emb_bf, wih_bf, b_lstm, xgates, BS, G4H, HID, G4H, G4H);

  // 4) Sequential LSTM scan: 512 x (recurrent GEMM + pointwise cell).
  //    W_hh (8 MiB bf16) stays L2-resident across all steps.
  init_state_kernel<<<(16 * HID + 255) / 256, 256, 0, stream>>>(h_bf, c_st);
  for (int t = 0; t < SEQ; ++t) {
    // gates[16,4096] = h_pad[16,1024] @ W_hh[1024,4096]
    gemm_bf16_wmma<<<dim3(G4H / 256, 1), 128, 0, stream>>>(
        h_bf, whh_bf, nullptr, gates, 16, G4H, HID, G4H, G4H);
    lstm_cell_kernel<<<(BATCH * HID + 255) / 256, 256, 0, stream>>>(
        xgates, gates, c_st, h_bf, hs_bf, t);
  }

  // 5) logits = hs @ w_out + b_out   [4096,1024]@[1024,32000] (HBM-bound: ~600MB)
  gemm_bf16_wmma<<<dim3(VOCAB / 256, BS / 16), 128, 0, stream>>>(
      hs_bf, wout_bf, b_out, logits, BS, VOCAB, HID, VOCAB, VOCAB);
}
